// SelfAttention_27925877358835
// MI455X (gfx1250) — matrix operand order repaired and checked by reading, not compiled, not run on hardware
//
#include <hip/hip_runtime.h>

typedef __attribute__((ext_vector_type(16))) _Float16 v16h;
typedef __attribute__((ext_vector_type(8)))  _Float16 v8h;
typedef __attribute__((ext_vector_type(4)))  _Float16 h4;
typedef __attribute__((ext_vector_type(8)))  float    v8f;
typedef __attribute__((ext_vector_type(4)))  float    f4;

constexpr int B = 4, H = 16, S = 2048, D = 128;
constexpr int QT  = 16;            // query rows per workgroup
constexpr int SP  = S + 4;         // padded f32 score row stride (dwords)
constexpr int PHP = S + 8;         // padded f16 prob row stride (halves, keeps 16B align)
constexpr size_t NE = (size_t)B * H * S * D;   // 16,777,216 elements per tensor
constexpr float SCALE = 0.08838834764831845f;  // 1/sqrt(128)

// ---------------- prepass 1: f32 -> f16 (row-major), 4 elem/thread ----------------
__global__ __launch_bounds__(256)
void cvt_f16_kernel(const float* __restrict__ in, _Float16* __restrict__ out)
{
    const size_t i = (size_t)blockIdx.x * 256 + threadIdx.x;
    const f4 v = __builtin_nontemporal_load((const f4*)in + i);   // f32 never read again
    ((h4*)out)[i] = __builtin_convertvector(v, h4);
}

// ---------------- prepass 2: V [bh][s][d] f32 -> Vt [bh][d][s] f16 ----------------
__global__ __launch_bounds__(256)
void transpose_v_kernel(const float* __restrict__ V, _Float16* __restrict__ Vt)
{
    __shared__ _Float16 tile[D][64 + 4];
    const int bh = blockIdx.x >> 5;            // 32 s-tiles of 64 per (b,h)
    const int s0 = (blockIdx.x & 31) * 64;
    const float* src = V + ((size_t)bh * S + s0) * D;
    for (int i = threadIdx.x; i < 64 * D; i += 256) {
        const int sl = i >> 7, d = i & 127;
        tile[d][sl] = (_Float16)__builtin_nontemporal_load(src + i);
    }
    __syncthreads();
    _Float16* dst = Vt + (size_t)bh * D * S + s0;
    for (int i = threadIdx.x; i < D * 64; i += 256) {
        const int d = i >> 6, sl = i & 63;
        dst[(size_t)d * S + sl] = tile[d][sl];
    }
}

static __device__ __forceinline__ v16h cat8(v8h lo, v8h hi) {
    return __builtin_shufflevector(lo, hi, 0,1,2,3,4,5,6,7,8,9,10,11,12,13,14,15);
}

// ---------------- fused attention: QK^T -> softmax -> P store -> P@V ----------------
__global__ __launch_bounds__(256)
void attn_fused_kernel(const _Float16* __restrict__ Qh,
                       const _Float16* __restrict__ Kh,
                       const _Float16* __restrict__ Vt,
                       float* __restrict__ Z,
                       float* __restrict__ P)
{
    __shared__ float    sc[QT * SP];     // 16 x 2052 f32 scores   (131 KB)
    __shared__ _Float16 ph[QT * PHP];    // 16 x 2056 f16 probs    (66 KB)
    __shared__ float    red[QT][16];
    __shared__ float    rowinv[QT];

    const int tid  = threadIdx.x;
    const int lane = tid & 31;
    const int wave = tid >> 5;
    const int m16  = lane & 15;          // row (A) or column (B/C) within 16x16 tile
    const int hi   = lane >> 4;          // half-wave select per ISA layout tables

    const int bh = blockIdx.x >> 7;      // S/QT = 128 query tiles per (b,h)
    const int q0 = (blockIdx.x & 127) * QT;

    // ---- A operand: Q tile rows q0..q0+15, f16, direct b128 loads in A layout ----
    v16h a[4];
    {
        const _Float16* qrow = Qh + ((size_t)bh * S + q0 + m16) * D;
        #pragma unroll
        for (int c = 0; c < 4; ++c) {
            const v8h lo = *(const v8h*)(qrow + c * 32 + hi * 8);
            const v8h up = *(const v8h*)(qrow + c * 32 + 16 + hi * 8);
            a[c] = cat8(lo, up);
        }
    }

    // ---- Phase 1: scores = (Q K^T) * scale.  Wave w owns keys [w*256, w*256+256).
    const _Float16* Kb = Kh + (size_t)bh * S * D;
    #pragma unroll 2
    for (int t = 0; t < 16; ++t) {
        const int nbase = wave * 256 + t * 16;
        const _Float16* krow = Kb + (size_t)(nbase + m16) * D;   // key row = column n
        v8f c = {};
        #pragma unroll
        for (int cc = 0; cc < 4; ++cc) {
            const v8h lo = *(const v8h*)(krow + cc * 32 + hi * 8);
            const v8h up = *(const v8h*)(krow + cc * 32 + 16 + hi * 8);
            c = __builtin_amdgcn_wmma_f32_16x16x32_f16(
                    false, a[cc], false, cat8(lo, up), (short)0, c, false, false);
        }
        #pragma unroll
        for (int r = 0; r < 8; ++r) {                 // C: dword r -> M = r+8*hi, N = m16
            sc[(r + 8 * hi) * SP + nbase + m16] = c[r] * SCALE;
        }
    }
    __syncthreads();

    // ---- Phase 2: row softmax (16 threads per row) ----
    {
        const int r = tid >> 4;
        const int k = tid & 15;
        float mx = -3.0e38f;
        for (int n = k; n < S; n += 16) mx = fmaxf(mx, sc[r * SP + n]);
        red[r][k] = mx;
        __syncthreads();
        float rmx = red[r][0];
        #pragma unroll
        for (int j = 1; j < 16; ++j) rmx = fmaxf(rmx, red[r][j]);
        float sum = 0.f;
        for (int n = k; n < S; n += 16) {
            const float e = __expf(sc[r * SP + n] - rmx);
            sc[r * SP + n] = e;
            sum += e;
        }
        __syncthreads();
        red[r][k] = sum;
        __syncthreads();
        if (k == 0) {
            float s = 0.f;
            #pragma unroll
            for (int j = 0; j < 16; ++j) s += red[r][j];
            rowinv[r] = 1.0f / s;
        }
    }
    __syncthreads();

    // ---- Phase 3: normalize; stream attention to HBM (non-temporal: keep L2 for K/Vt);
    //      also deposit f16 probs in LDS as phase-4 A operand source. ----
    {
        float* Pb = P + ((size_t)bh * S + q0) * S;
        for (int r = 0; r < QT; ++r) {
            const float inv = rowinv[r];
            for (int n = tid; n < S; n += 256) {
                const float p = sc[r * SP + n] * inv;
                __builtin_nontemporal_store(p, &Pb[(size_t)r * S + n]);
                ph[r * PHP + n] = (_Float16)p;
            }
        }
    }
    __syncthreads();

    // ---- Phase 4: Z = P @ V.  Wave w owns output columns [w*16, w*16+16).
    //      A: 2x ds_load_b128 from f16 probs; B: 2x global_load_b128 from Vt (L2-hot).
    v8f zc = {};
    {
        const int col = wave * 16 + m16;
        const _Float16* vrow = Vt + ((size_t)bh * D + col) * S;  // Vt[n][k], k contiguous
        const _Float16* prow = ph + m16 * PHP;
        #pragma unroll 4
        for (int kc = 0; kc < S / 32; ++kc) {
            const v8h plo = *(const v8h*)(prow + kc * 32 + hi * 8);
            const v8h pup = *(const v8h*)(prow + kc * 32 + 16 + hi * 8);
            const v8h vlo = *(const v8h*)(vrow + kc * 32 + hi * 8);
            const v8h vup = *(const v8h*)(vrow + kc * 32 + 16 + hi * 8);
            zc = __builtin_amdgcn_wmma_f32_16x16x32_f16(
                     false, cat8(plo, pup), false, cat8(vlo, vup),
                     (short)0, zc, false, false);
        }
    }

    // ---- write Z tile ----
    {
        float* Zb = Z + ((size_t)bh * S + q0) * D;
        const int col = wave * 16 + m16;
        #pragma unroll
        for (int r = 0; r < 8; ++r)
            Zb[(size_t)(r + 8 * hi) * D + col] = zc[r];
    }
}

extern "C" void kernel_launch(void* const* d_in, const int* in_sizes, int n_in,
                              void* d_out, int out_size, void* d_ws, size_t ws_size,
                              hipStream_t stream) {
    const float* Q = (const float*)d_in[0];
    const float* K = (const float*)d_in[1];
    const float* V = (const float*)d_in[2];
    float* Z = (float*)d_out;
    float* P = (float*)d_out + NE;           // attention follows Z in d_out

    _Float16* Qh = (_Float16*)d_ws;          // 32 MB
    _Float16* Kh = Qh + NE;                  // 32 MB
    _Float16* Vt = Kh + NE;                  // 32 MB (transposed V)

    const int cvtBlocks = (int)(NE / (256 * 4));
    cvt_f16_kernel<<<cvtBlocks, 256, 0, stream>>>(Q, Qh);
    cvt_f16_kernel<<<cvtBlocks, 256, 0, stream>>>(K, Kh);
    transpose_v_kernel<<<B * H * (S / 64), 256, 0, stream>>>(V, Vt);

    attn_fused_kernel<<<B * H * (S / QT), 256, 0, stream>>>(Qh, Kh, Vt, Z, P);
}
